// GraphGPS_22892175688267
// MI455X (gfx1250) — matrix operand order repaired and checked by reading, not compiled, not run on hardware
//
#include <hip/hip_runtime.h>
#include <math.h>

// ---------------- problem constants (match reference) ----------------
#define N_NODES   6144
#define C_DIM     256
#define G_GRAPHS  256
#define NP_ATOMS  24
#define E_EDGES   147456
#define E_DIM     69
#define E_PAD     96          // eattr row stride: 3 K-tiles, zero padded
#define NT_TYPES  100
#define L_LAYERS  3

typedef float        v8f  __attribute__((ext_vector_type(8)));
typedef __bf16       v16bf __attribute__((ext_vector_type(16)));
typedef unsigned int v8u  __attribute__((ext_vector_type(8)));

// round-half-up fp32 -> bf16 (1 add + shift; ties differ from RNE only at 2^-16)
__device__ __forceinline__ unsigned short f2bf(float f) {
  return (unsigned short)((__float_as_uint(f) + 0x8000u) >> 16);
}
__device__ __forceinline__ unsigned int pack2bf(float a, float b) {
  unsigned int lo = (__float_as_uint(a) + 0x8000u) >> 16;
  unsigned int hi = (__float_as_uint(b) + 0x8000u) & 0xFFFF0000u;
  return lo | hi;
}
__device__ __forceinline__ float silu_f(float v) { return v / (1.f + __expf(-v)); }

// =====================================================================
// bf16-WMMA GEMM: Out[M,Nc] = act((A+A2)[M,Kpad(valid K)] @ W[K,Nc] + bias + resid)
// SCATTER (GINE edge conv): msg = relu(x[src]+gemm+bias); LDS pre-reduce per
//   (local dst, col), then one global atomicAdd per (dst, col).
// 256 threads = 8 wave32 waves; tile 64x64x32; wave: m-tile w&3, n-tiles (w>>2)*2+{0,1}
// =====================================================================
#define TM 64
#define TN 64
#define TK 32

template<int ACT, bool SCATTER, bool HAS_A2>
__global__ __launch_bounds__(256)
void gps_gemm_kernel(const float* __restrict__ A,  const float* __restrict__ A2,
                     const float* __restrict__ W,  const float* __restrict__ bias,
                     const float* __restrict__ resid, float* __restrict__ Out,
                     int M, int K, int Kpad, int Ncols,
                     const int* __restrict__ esrc, const int* __restrict__ edst,
                     const float* __restrict__ xsrc, float* __restrict__ aggr)
{
  __shared__ unsigned short As[TM][TK + 2];   // [m][k] bf16
  __shared__ unsigned short Bs[TN][TK + 2];   // [n][k] bf16 (W transposed in LDS)

  const int tid    = threadIdx.x;
  const int lane   = tid & 31;
  const int wave   = tid >> 5;
  const int blockM = blockIdx.y * TM;
  const int blockN = blockIdx.x * TN;
  const int mt     = wave & 3;
  const int ntBase = (wave >> 2) * 2;
  const int half   = lane >> 4;     // K-half selector
  const int sub    = lane & 15;     // row (A frag) / col (B/D frag)

  const bool fullM = (blockM + TM <= M) && ((Kpad & 3) == 0);
  const bool fullN = (blockN + TN <= Ncols) && ((Ncols & 3) == 0);

  v8f accs[2];
#pragma unroll
  for (int i = 0; i < 2; ++i)
#pragma unroll
    for (int r = 0; r < 8; ++r) accs[i][r] = 0.f;

  for (int k0 = 0; k0 < Kpad; k0 += TK) {
    // ------------- stage A tile (64x32) -------------
    if (fullM && (k0 + TK <= Kpad)) {
      // vectorized: thread -> 8 consecutive k of one row (2x b128 loads)
      int r  = tid >> 2;
      int kk = (tid & 3) * 8;
      const float* pA = A + (size_t)(blockM + r) * Kpad + k0 + kk;
      float4 a0 = *(const float4*)pA;
      float4 a1 = *(const float4*)(pA + 4);
      if (HAS_A2) {
        const float* pA2 = A2 + (size_t)(blockM + r) * Kpad + k0 + kk;
        float4 c0 = *(const float4*)pA2;
        float4 c1 = *(const float4*)(pA2 + 4);
        a0.x += c0.x; a0.y += c0.y; a0.z += c0.z; a0.w += c0.w;
        a1.x += c1.x; a1.y += c1.y; a1.z += c1.z; a1.w += c1.w;
      }
      unsigned int* d32 = (unsigned int*)&As[r][kk];
      d32[0] = pack2bf(a0.x, a0.y);
      d32[1] = pack2bf(a0.z, a0.w);
      d32[2] = pack2bf(a1.x, a1.y);
      d32[3] = pack2bf(a1.z, a1.w);
      if (k0 + TK < Kpad) __builtin_prefetch(pA + TK, 0, 0);
    } else {
#pragma unroll
      for (int t = 0; t < 8; ++t) {
        int linear = tid * 8 + t;
        int r = linear >> 5, kk = linear & 31;
        int gm = blockM + r, gk = k0 + kk;
        float a = 0.f;
        if (gm < M && gk < Kpad) {
          a = A[(size_t)gm * Kpad + gk];
          if (HAS_A2) a += A2[(size_t)gm * Kpad + gk];
        }
        As[r][kk] = f2bf(a);
      }
    }
    // ------------- stage W tile (32x64, transposed into Bs[n][k]) -------------
    if (fullN && (k0 + TK <= K)) {
      int n  = (tid & 7) * 8;
      int kk = tid >> 3;
      const float* pW = W + (size_t)(k0 + kk) * Ncols + blockN + n;
      float4 b0 = *(const float4*)pW;
      float4 b1 = *(const float4*)(pW + 4);
      Bs[n + 0][kk] = f2bf(b0.x);
      Bs[n + 1][kk] = f2bf(b0.y);
      Bs[n + 2][kk] = f2bf(b0.z);
      Bs[n + 3][kk] = f2bf(b0.w);
      Bs[n + 4][kk] = f2bf(b1.x);
      Bs[n + 5][kk] = f2bf(b1.y);
      Bs[n + 6][kk] = f2bf(b1.z);
      Bs[n + 7][kk] = f2bf(b1.w);
      if (k0 + TK < K) __builtin_prefetch(pW + (size_t)TK * Ncols, 0, 0);
    } else {
#pragma unroll
      for (int t = 0; t < 8; ++t) {
        int linear = tid * 8 + t;
        int n = linear & 63, kk = linear >> 6;
        int gn = blockN + n, gk = k0 + kk;
        float b = 0.f;
        if (gn < Ncols && gk < K) b = W[(size_t)gk * Ncols + gn];
        Bs[n][kk] = f2bf(b);
      }
    }
    __syncthreads();

    // ---- A fragment: 16-bit A 16x32 VGPR layout (ISA 7.12.2) ----
    v8u au;
#pragma unroll
    for (int p = 0; p < 4; ++p) {
      au[p]     = *(const unsigned int*)&As[mt * 16 + sub][half * 8 + 2 * p];
      au[p + 4] = *(const unsigned int*)&As[mt * 16 + sub][16 + half * 8 + 2 * p];
    }
    v16bf av = __builtin_bit_cast(v16bf, au);

#pragma unroll
    for (int nn = 0; nn < 2; ++nn) {
      int nt = ntBase + nn;
      v8u bu;
#pragma unroll
      for (int p = 0; p < 8; ++p)
        bu[p] = *(const unsigned int*)&Bs[nt * 16 + sub][half * 8 + 2 * p + (p & 4) * 2];
      v16bf bv = __builtin_bit_cast(v16bf, bu);
      accs[nn] = __builtin_amdgcn_wmma_f32_16x16x32_bf16(
          false, av, false, bv, (short)0, accs[nn], false, false);
    }
    __syncthreads();
  }

  // ------------- epilogue -------------
  if constexpr (SCATTER) {
    // every 64-row tile lies inside one graph (576 edges/graph = 9 tiles):
    // pre-reduce the 24-way dst collisions in LDS, then one global atomic each.
    __shared__ float acc_lds[NP_ATOMS][TN];
    for (int i = tid; i < NP_ATOMS * TN; i += 256)
      ((float*)acc_lds)[i] = 0.f;
    __syncthreads();
#pragma unroll
    for (int r = 0; r < 8; ++r) {
      int row = blockM + mt * 16 + half * 8 + r;
      int s  = esrc[row];
      int li = edst[row] % NP_ATOMS;          // local dst within graph
      const float* xr = xsrc + (size_t)s * C_DIM;
#pragma unroll
      for (int nn = 0; nn < 2; ++nn) {
        int col = blockN + (ntBase + nn) * 16 + sub;
        float v = accs[nn][r] + bias[col];
        atomicAdd(&acc_lds[li][col - blockN], fmaxf(xr[col] + v, 0.f)); // ds_add_f32
      }
    }
    __syncthreads();
    int gbase = (blockM / (NP_ATOMS * NP_ATOMS)) * NP_ATOMS;  // graph * 24
    for (int i = tid; i < NP_ATOMS * TN; i += 256) {
      int li = i >> 6, c = i & 63;
      atomicAdd(&aggr[(size_t)(gbase + li) * C_DIM + blockN + c], acc_lds[li][c]);
    }
  } else {
#pragma unroll
    for (int nn = 0; nn < 2; ++nn) {
      int nt = ntBase + nn;
#pragma unroll
      for (int r = 0; r < 8; ++r) {
        int row = blockM + mt * 16 + half * 8 + r;
        int col = blockN + nt * 16 + sub;
        if (row < M && col < Ncols) {
          float v = accs[nn][r];
          if (bias) v += bias[col];
          if (resid) v += resid[(size_t)row * Ncols + col];
          if (ACT == 1) v = fmaxf(v, 0.f);
          else if (ACT == 2) v = silu_f(v);
          Out[(size_t)row * Ncols + col] = v;
        }
      }
    }
  }
}

// ---------------- elementwise / small kernels ----------------
__global__ void gps_fill_zero(float* p, int n) {
  int i = blockIdx.x * 256 + threadIdx.x;
  if (i < n) p[i] = 0.f;
}

__global__ void gps_add(const float* a, const float* b, float* y, int n) {
  int i = blockIdx.x * 256 + threadIdx.x;
  if (i < n) y[i] = a[i] + b[i];
}

// xin[N,512] = concat(type_emb[atom_types], sinusoidal time emb)
__global__ void gps_build_xin(const float* __restrict__ t,
                              const float* __restrict__ type_emb,
                              const int* __restrict__ atom_types,
                              float* __restrict__ xin, int n)
{
  int i = blockIdx.x * 256 + threadIdx.x;
  if (i >= n) return;
  int row = i >> 9, col = i & 511;
  float v;
  if (col < 256) {
    v = type_emb[(size_t)atom_types[row] * 256 + col];
  } else {
    int cc = col - 256;
    int g = row / NP_ATOMS;
    float tv = t[g];
    int idx = (cc < 128) ? cc : cc - 128;
    float freq = __expf(-9.210340372f * (float)idx / 127.f);  // log(1e4)
    v = (cc < 128) ? __sinf(tv * freq) : __cosf(tv * freq);
  }
  xin[i] = v;
}

// eattr[E,96] = [sin(fd*2pi*f) (30) | cos (30) | lattice[g] (9) | zero pad (27)]
__global__ void gps_build_eattr(const float* __restrict__ frac,
                                const float* __restrict__ lattice,
                                const int* __restrict__ esrc,
                                const int* __restrict__ edst,
                                const int* __restrict__ n2g,
                                float* __restrict__ eattr, int n)
{
  int i = blockIdx.x * 256 + threadIdx.x;
  if (i >= n) return;
  int e = i / E_PAD, col = i - e * E_PAD;
  float v = 0.f;
  if (col < 60) {
    int s = esrc[e], d = edst[e];
    int c30 = col % 30;
    int dim = c30 / 10, f = c30 - dim * 10;
    float fd = frac[(size_t)d * 3 + dim] - frac[(size_t)s * 3 + dim];
    fd -= floorf(fd);                             // mod 1.0
    float arg = fd * 6.283185307f * (float)f;
    v = (col < 30) ? __sinf(arg) : __cosf(arg);
  } else if (col < E_DIM) {
    int g = n2g[esrc[e]];
    v = lattice[(size_t)g * 9 + (col - 60)];
  }
  eattr[i] = v;
}

// BatchNorm stats over rows: one block per channel
__global__ __launch_bounds__(256)
void gps_bn_stats(const float* __restrict__ X, int rows, int C,
                  float* __restrict__ mean, float* __restrict__ rstd)
{
  int c = blockIdx.x;
  float s = 0.f, s2 = 0.f;
  for (int r = threadIdx.x; r < rows; r += 256) {
    float v = X[(size_t)r * C + c];
    s += v; s2 += v * v;
  }
  __shared__ float sh[256], sh2[256];
  sh[threadIdx.x] = s; sh2[threadIdx.x] = s2;
  __syncthreads();
  for (int st = 128; st > 0; st >>= 1) {
    if (threadIdx.x < st) { sh[threadIdx.x] += sh[threadIdx.x + st];
                            sh2[threadIdx.x] += sh2[threadIdx.x + st]; }
    __syncthreads();
  }
  if (threadIdx.x == 0) {
    float m   = sh[0] / (float)rows;
    float var = sh2[0] / (float)rows - m * m;
    mean[c] = m;
    rstd[c] = rsqrtf(var + 1e-5f);
  }
}

__global__ void gps_bn_apply(const float* __restrict__ X, float* __restrict__ Y,
                             const float* __restrict__ mean, const float* __restrict__ rstd,
                             const float* __restrict__ g, const float* __restrict__ b, int n)
{
  int i = blockIdx.x * 256 + threadIdx.x;
  if (i >= n) return;
  int c = i & (C_DIM - 1);
  Y[i] = (X[i] - mean[c]) * rstd[c] * g[c] + b[c];
}

// dense 24x24 multihead attention, one block per (graph, head)
__global__ __launch_bounds__(256)
void gps_attn(const float* __restrict__ qkv, float* __restrict__ o)
{
  __shared__ float q[NP_ATOMS][64], k[NP_ATOMS][64], v[NP_ATOMS][64];
  __shared__ float S[NP_ATOMS][NP_ATOMS];
  int g = blockIdx.x >> 2, h = blockIdx.x & 3;
  int tid = threadIdx.x;

  for (int i = tid; i < NP_ATOMS * 64; i += 256) {
    int p = i >> 6, d = i & 63;
    size_t base = (size_t)(g * NP_ATOMS + p) * 768;
    q[p][d] = qkv[base + h * 64 + d];
    k[p][d] = qkv[base + 256 + h * 64 + d];
    v[p][d] = qkv[base + 512 + h * 64 + d];
  }
  __syncthreads();
  for (int s = tid; s < NP_ATOMS * NP_ATOMS; s += 256) {
    int qi = s / NP_ATOMS, kj = s - qi * NP_ATOMS;
    float acc = 0.f;
#pragma unroll 8
    for (int d = 0; d < 64; ++d) acc += q[qi][d] * k[kj][d];
    S[qi][kj] = acc * 0.125f;                    // 1/sqrt(64)
  }
  __syncthreads();
  if (tid < NP_ATOMS) {
    float mx = -1e30f;
    for (int j = 0; j < NP_ATOMS; ++j) mx = fmaxf(mx, S[tid][j]);
    float sum = 0.f;
    for (int j = 0; j < NP_ATOMS; ++j) { float e = __expf(S[tid][j] - mx); S[tid][j] = e; sum += e; }
    float inv = 1.f / sum;
    for (int j = 0; j < NP_ATOMS; ++j) S[tid][j] *= inv;
  }
  __syncthreads();
  for (int i = tid; i < NP_ATOMS * 64; i += 256) {
    int p = i >> 6, d = i & 63;
    float acc = 0.f;
#pragma unroll
    for (int j = 0; j < NP_ATOMS; ++j) acc += S[p][j] * v[j][d];
    o[(size_t)(g * NP_ATOMS + p) * C_DIM + h * 64 + d] = acc;
  }
}

// LayerNorm over feature dim, one block (=256 ch) per row
__global__ __launch_bounds__(256)
void gps_layernorm(const float* __restrict__ X, float* __restrict__ Y,
                   const float* __restrict__ g, const float* __restrict__ b)
{
  int row = blockIdx.x, c = threadIdx.x;
  float v = X[(size_t)row * C_DIM + c];
  __shared__ float sh[256], sh2[256];
  sh[c] = v; sh2[c] = v * v;
  __syncthreads();
  for (int st = 128; st > 0; st >>= 1) {
    if (c < st) { sh[c] += sh[c + st]; sh2[c] += sh2[c + st]; }
    __syncthreads();
  }
  float m = sh[0] * (1.f / 256.f);
  float var = sh2[0] * (1.f / 256.f) - m * m;
  float rs = rsqrtf(var + 1e-5f);
  Y[(size_t)row * C_DIM + c] = (v - m) * rs * g[c] + b[c];
}

// scatter-mean pooling, one block per graph
__global__ __launch_bounds__(256)
void gps_pool(const float* __restrict__ hn, float* __restrict__ gf)
{
  int g = blockIdx.x, c = threadIdx.x;
  float s = 0.f;
#pragma unroll
  for (int p = 0; p < NP_ATOMS; ++p)
    s += hn[(size_t)(g * NP_ATOMS + p) * C_DIM + c];
  gf[(size_t)g * C_DIM + c] = s * (1.f / (float)NP_ATOMS);
}

// ---------------- host-side launch helpers ----------------
static void launch_gemm(int act, const float* A, const float* W,
                        const float* bias, const float* resid, float* Out,
                        int M, int K, int Nc, hipStream_t st)
{
  dim3 grid((Nc + TN - 1) / TN, (M + TM - 1) / TM), block(256);
  switch (act) {
    case 1: gps_gemm_kernel<1, false, false><<<grid, block, 0, st>>>(A, nullptr, W, bias, resid, Out, M, K, K, Nc, nullptr, nullptr, nullptr, nullptr); break;
    case 2: gps_gemm_kernel<2, false, false><<<grid, block, 0, st>>>(A, nullptr, W, bias, resid, Out, M, K, K, Nc, nullptr, nullptr, nullptr, nullptr); break;
    default: gps_gemm_kernel<0, false, false><<<grid, block, 0, st>>>(A, nullptr, W, bias, resid, Out, M, K, K, Nc, nullptr, nullptr, nullptr, nullptr); break;
  }
}

static void launch_gemm_add2_silu(const float* A, const float* A2, const float* W,
                                  const float* bias, float* Out,
                                  int M, int K, int Nc, hipStream_t st)
{
  dim3 grid((Nc + TN - 1) / TN, (M + TM - 1) / TM), block(256);
  gps_gemm_kernel<2, false, true><<<grid, block, 0, st>>>(A, A2, W, bias, nullptr, Out, M, K, K, Nc, nullptr, nullptr, nullptr, nullptr);
}

static void launch_edge_gemm(const float* eattr, const float* W, const float* bias,
                             const int* esrc, const int* edst, const float* x,
                             float* aggr, hipStream_t st)
{
  dim3 grid((C_DIM + TN - 1) / TN, (E_EDGES + TM - 1) / TM), block(256);
  gps_gemm_kernel<0, true, false><<<grid, block, 0, st>>>(eattr, nullptr, W, bias, nullptr,
                                                          nullptr, E_EDGES, E_DIM, E_PAD, C_DIM,
                                                          esrc, edst, x, aggr);
}

extern "C" void kernel_launch(void* const* d_in, const int* in_sizes, int n_in,
                              void* d_out, int out_size, void* d_ws, size_t ws_size,
                              hipStream_t stream)
{
  (void)in_sizes; (void)n_in; (void)out_size; (void)ws_size;
  // -------- inputs (setup_inputs order) --------
  const float* t        = (const float*)d_in[0];
  const float* frac     = (const float*)d_in[1];
  const float* lattice  = (const float*)d_in[2];
  const float* type_emb = (const float*)d_in[3];
  const float* naW      = (const float*)d_in[4];
  const float* nab      = (const float*)d_in[5];
  const float* geW      = (const float*)d_in[6];
  const float* geb      = (const float*)d_in[7];
  const float* gW1      = (const float*)d_in[8];
  const float* gb1      = (const float*)d_in[9];
  const float* gW2      = (const float*)d_in[10];
  const float* gb2      = (const float*)d_in[11];
  const float* aiW      = (const float*)d_in[12];
  const float* aib      = (const float*)d_in[13];
  const float* aoW      = (const float*)d_in[14];
  const float* aob      = (const float*)d_in[15];
  const float* mW1      = (const float*)d_in[16];
  const float* mb1      = (const float*)d_in[17];
  const float* mW2      = (const float*)d_in[18];
  const float* mb2      = (const float*)d_in[19];
  const float* bn1g     = (const float*)d_in[20];
  const float* bn1b     = (const float*)d_in[21];
  const float* bn2g     = (const float*)d_in[22];
  const float* bn2b     = (const float*)d_in[23];
  const float* bn3g     = (const float*)d_in[24];
  const float* bn3b     = (const float*)d_in[25];
  const float* lng      = (const float*)d_in[26];
  const float* lnb      = (const float*)d_in[27];
  const float* troW1    = (const float*)d_in[28];
  const float* trob1    = (const float*)d_in[29];
  const float* troW2    = (const float*)d_in[30];
  const float* trob2    = (const float*)d_in[31];
  const float* latW1    = (const float*)d_in[32];
  const float* latb1    = (const float*)d_in[33];
  const float* latW2    = (const float*)d_in[34];
  const float* latb2    = (const float*)d_in[35];
  const float* latW3    = (const float*)d_in[36];
  const float* fcW1     = (const float*)d_in[37];
  const float* fcb1     = (const float*)d_in[38];
  const float* fcW2     = (const float*)d_in[39];
  const float* fcb2     = (const float*)d_in[40];
  const float* fcW3     = (const float*)d_in[41];
  const int*   atom_t   = (const int*)d_in[42];
  const int*   n2g      = (const int*)d_in[43];
  const int*   esrc     = (const int*)d_in[44];
  const int*   edst     = (const int*)d_in[45];

  // -------- workspace carve-up (floats) --------
  float* ws = (float*)d_ws;
  size_t off = 0;
  auto alloc = [&](size_t n) { float* p = ws + off; off += n; return p; };
  float* eattr = alloc((size_t)E_EDGES * E_PAD);   // 14.2M (zero-padded K)
  float* big   = alloc((size_t)N_NODES * 768);     // xin / qkv / mlp-hidden
  float* x     = alloc((size_t)N_NODES * C_DIM);
  float* aggr  = alloc((size_t)N_NODES * C_DIM);
  float* t1    = alloc((size_t)N_NODES * C_DIM);
  float* h1    = alloc((size_t)N_NODES * C_DIM);
  float* h2    = alloc((size_t)N_NODES * C_DIM);
  float* sbuf  = alloc((size_t)N_NODES * C_DIM);
  float* hn    = alloc((size_t)N_NODES * C_DIM);
  float* gf    = alloc((size_t)G_GRAPHS * C_DIM);
  float* hd128 = alloc((size_t)N_NODES * 128);
  float* hd64  = alloc((size_t)N_NODES * 64);
  float* mean  = alloc(C_DIM);
  float* rstd  = alloc(C_DIM);

  const int NC = N_NODES * C_DIM;
  dim3 b256(256);
  auto gridFor = [](int n) { return dim3((n + 255) / 256); };

  // -------- input embeddings --------
  gps_build_xin<<<gridFor(N_NODES * 512), b256, 0, stream>>>(t, type_emb, atom_t, big, N_NODES * 512);
  launch_gemm(0, big, naW, nab, nullptr, x, N_NODES, 512, C_DIM, stream);
  gps_build_eattr<<<gridFor(E_EDGES * E_PAD), b256, 0, stream>>>(frac, lattice, esrc, edst, n2g, eattr, E_EDGES * E_PAD);

  // -------- GPS layers --------
  for (int l = 0; l < L_LAYERS; ++l) {
    const float* geWl = geW + (size_t)l * E_DIM * C_DIM;
    const float* gebl = geb + (size_t)l * C_DIM;
    const float* gW1l = gW1 + (size_t)l * C_DIM * C_DIM;
    const float* gb1l = gb1 + (size_t)l * C_DIM;
    const float* gW2l = gW2 + (size_t)l * C_DIM * C_DIM;
    const float* gb2l = gb2 + (size_t)l * C_DIM;
    const float* aiWl = aiW + (size_t)l * C_DIM * 768;
    const float* aibl = aib + (size_t)l * 768;
    const float* aoWl = aoW + (size_t)l * C_DIM * C_DIM;
    const float* aobl = aob + (size_t)l * C_DIM;
    const float* mW1l = mW1 + (size_t)l * C_DIM * 512;
    const float* mb1l = mb1 + (size_t)l * 512;
    const float* mW2l = mW2 + (size_t)l * 512 * C_DIM;
    const float* mb2l = mb2 + (size_t)l * C_DIM;

    // GINEConv: aggr = scatter-add(relu(x[src] + eattr@We + be))
    gps_fill_zero<<<gridFor(NC), b256, 0, stream>>>(aggr, NC);
    launch_edge_gemm(eattr, geWl, gebl, esrc, edst, x, aggr, stream);
    // t1 = silu((x+aggr)@W1+b1) ; h1raw = t1@W2+b2 + x ; h1 = BN1
    launch_gemm_add2_silu(x, aggr, gW1l, gb1l, t1, N_NODES, C_DIM, C_DIM, stream);
    launch_gemm(0, t1, gW2l, gb2l, x, h1, N_NODES, C_DIM, C_DIM, stream);
    gps_bn_stats<<<C_DIM, b256, 0, stream>>>(h1, N_NODES, C_DIM, mean, rstd);
    gps_bn_apply<<<gridFor(NC), b256, 0, stream>>>(h1, h1, mean, rstd, bn1g + l * C_DIM, bn1b + l * C_DIM, NC);

    // attention: qkv -> per-(g,h) softmax -> out proj (+x) -> BN2
    launch_gemm(0, x, aiWl, aibl, nullptr, big, N_NODES, C_DIM, 768, stream);
    gps_attn<<<G_GRAPHS * 4, b256, 0, stream>>>(big, t1);
    launch_gemm(0, t1, aoWl, aobl, x, h2, N_NODES, C_DIM, C_DIM, stream);
    gps_bn_stats<<<C_DIM, b256, 0, stream>>>(h2, N_NODES, C_DIM, mean, rstd);
    gps_bn_apply<<<gridFor(NC), b256, 0, stream>>>(h2, h2, mean, rstd, bn2g + l * C_DIM, bn2b + l * C_DIM, NC);

    // combine + MLP + BN3 -> new x
    gps_add<<<gridFor(NC), b256, 0, stream>>>(h1, h2, sbuf, NC);
    launch_gemm(1, sbuf, mW1l, mb1l, nullptr, big, N_NODES, C_DIM, 512, stream);
    launch_gemm(0, big, mW2l, mb2l, sbuf, h1, N_NODES, 512, C_DIM, stream);
    gps_bn_stats<<<C_DIM, b256, 0, stream>>>(h1, N_NODES, C_DIM, mean, rstd);
    gps_bn_apply<<<gridFor(NC), b256, 0, stream>>>(h1, x, mean, rstd, bn3g + l * C_DIM, bn3b + l * C_DIM, NC);
  }

  // -------- final norm, pooling, heads --------
  gps_layernorm<<<N_NODES, b256, 0, stream>>>(x, hn, lng, lnb);
  gps_pool<<<G_GRAPHS, b256, 0, stream>>>(hn, gf);

  float* out_types = (float*)d_out;                               // [N,100]
  float* out_lat   = out_types + (size_t)N_NODES * NT_TYPES;      // [G,9]
  float* out_frac  = out_lat + (size_t)G_GRAPHS * 9;              // [N,3]

  // types_pred = silu(hn@troW1+b1)@troW2+b2
  launch_gemm(2, hn, troW1, trob1, nullptr, hd128, N_NODES, C_DIM, 128, stream);
  launch_gemm(0, hd128, troW2, trob2, nullptr, out_types, N_NODES, 128, NT_TYPES, stream);
  // lattice_pred = silu(silu(gf@W1+b1)@W2+b2)@W3
  launch_gemm(2, gf, latW1, latb1, nullptr, hd128, G_GRAPHS, C_DIM, 128, stream);
  launch_gemm(2, hd128, latW2, latb2, nullptr, hd64, G_GRAPHS, 128, 64, stream);
  launch_gemm(0, hd64, latW3, nullptr, nullptr, out_lat, G_GRAPHS, 64, 9, stream);
  // frac_pred = silu(silu(hn@W1+b1)@W2+b2)@W3
  launch_gemm(2, hn, fcW1, fcb1, nullptr, hd128, N_NODES, C_DIM, 128, stream);
  launch_gemm(2, hd128, fcW2, fcb2, nullptr, hd64, N_NODES, 128, 64, stream);
  launch_gemm(0, hd64, fcW3, nullptr, nullptr, out_frac, N_NODES, 64, 3, stream);
}
